// TransformerEncoder_1159641170486
// MI455X (gfx1250) — compile-verified, matrix operand order checked by reading, not run on hardware
//
#include <hip/hip_runtime.h>
#include <hip/hip_bf16.h>

// ---------------------------------------------------------------------------
// Transformer encoder layer for MI455X (gfx1250, wave32, WMMA + TDM).
// GEMM-shaped math: v_wmma_f32_16x16x32_bf16 (bf16 in, f32 acc).
// B panels are staged into LDS by the Tensor Data Mover (tensor_load_to_lds,
// double-buffered, s_wait_tensorcnt) and shared by all 8 waves of the block.
// softmax / LayerNorm / GELU stay in f32 VALU.
// ---------------------------------------------------------------------------

typedef __attribute__((ext_vector_type(16))) __bf16       v16bf;
typedef __attribute__((ext_vector_type(8)))  __bf16       v8bf;
typedef __attribute__((ext_vector_type(8)))  float        v8f;
typedef __attribute__((ext_vector_type(4)))  unsigned int v4u;
typedef __attribute__((ext_vector_type(8)))  int          v8i;
typedef __attribute__((ext_vector_type(4)))  int          v4i;

#define DEV __device__ __forceinline__

constexpr int BB  = 2;
constexpr int S   = 2048;
constexpr int H   = 1024;
constexpr int NH  = 16;
constexpr int DK  = 64;
constexpr int FFN = 4096;
constexpr int M   = BB * S;   // 4096 rows

// ---- WMMA fragment helpers -------------------------------------------------
// 16-bit A-matrix 16x32 layout (ISA 7.12.2): lane l holds row (l&15); VGPRs
// 0..3 hold K = 8*(l>>4)+[0..7], VGPRs 4..7 hold K = 16 + 8*(l>>4)+[0..7]
// => per lane two contiguous 16B chunks. B operand kept transposed ([N,K])
// so it loads with the identical pattern (works for global and LDS).
DEV v16bf load_frag(const __bf16* __restrict__ base, int ld) {
  const int lane = threadIdx.x & 31;
  const int row  = lane & 15;
  const int half = lane >> 4;
  const __bf16* p = base + (size_t)row * ld + half * 8;
  v8bf lo = *reinterpret_cast<const v8bf*>(p);
  v8bf hi = *reinterpret_cast<const v8bf*>(p + 16);
  v16bf f;
#pragma unroll
  for (int i = 0; i < 8; ++i) { f[i] = lo[i]; f[i + 8] = hi[i]; }
  return f;
}

DEV v8f wmma_bf16(v16bf a, v16bf b, v8f c) {
  // 8 args: (neg_a, A, neg_b, B, c_mod, C, reuse_a, reuse_b)
  return __builtin_amdgcn_wmma_f32_16x16x32_bf16(false, a, false, b,
                                                 (short)0, c, false, false);
}

// ---- Tensor Data Mover: async-load a [128 rows x 32 k] bf16 panel ----------
// D# per CDNA5 ISA. 2D tensor: Bt is [N,K] bf16 row-major,
// tile = tile_dim0(=32, contiguous K) x tile_dim1(=128 rows), dim0_stride = K.
// Group0: count=1 | lds_addr | global_addr | type=2. Groups 2/3 unused (zero).
DEV void tdm_load_panel(const __bf16* gsrc, __bf16* ldsDst, int K, int N) {
  const unsigned long long ga = (unsigned long long)(uintptr_t)gsrc;
  const unsigned int lds = (unsigned int)(uintptr_t)ldsDst; // addr[31:0] = LDS offset
  v4u g0;
  g0[0] = 1u;                                   // count=1 (valid user D#)
  g0[1] = lds;                                  // lds_addr  [63:32]
  g0[2] = (unsigned int)ga;                     // global_addr[31:0]
  g0[3] = (unsigned int)(ga >> 32) | (2u << 30);// global_addr[56:32] | type=2
  v8i g1;
  g1[0] = 0x00010000;                           // data_size=1 (2 bytes/elem)
  g1[1] = (int)(((unsigned)K & 0xFFFFu) << 16); // tensor_dim0[15:0] @ [63:48]
  g1[2] = (int)((((unsigned)K >> 16) & 0xFFFFu) |
                (((unsigned)N & 0xFFFFu) << 16)); // dim0 hi | tensor_dim1 lo
  g1[3] = (int)((((unsigned)N >> 16) & 0xFFFFu) |
                (32u << 16));                   // tensor_dim1 hi | tile_dim0=32
  g1[4] = 128;                                  // tile_dim1=128 (tile_dim2=0)
  g1[5] = K;                                    // tensor_dim0_stride [207:160]
  g1[6] = 0;
  g1[7] = 0;                                    // tensor_dim1_stride unused
  v4i gz = {0, 0, 0, 0};
#if __clang_major__ >= 23
  v8i gz8 = {0, 0, 0, 0, 0, 0, 0, 0};
  __builtin_amdgcn_tensor_load_to_lds(g0, g1, gz, gz, gz8, 0);
#else
  __builtin_amdgcn_tensor_load_to_lds(g0, g1, gz, gz, 0);
#endif
}

// ---- elementwise cast / transpose ------------------------------------------
__global__ __launch_bounds__(256) void cast_bf16(const float* __restrict__ in,
                                                 __bf16* __restrict__ out, int n) {
  for (int i = blockIdx.x * 256 + threadIdx.x; i < n; i += gridDim.x * 256)
    out[i] = (__bf16)in[i];
}

// in: [R,C] f32 row-major -> out: [C,R] bf16 row-major
__global__ __launch_bounds__(256) void transpose_bf16(const float* __restrict__ in,
                                                      __bf16* __restrict__ out,
                                                      int R, int C) {
  int n = R * C;
  for (int i = blockIdx.x * 256 + threadIdx.x; i < n; i += gridDim.x * 256) {
    int c = i / R, r = i - c * R;
    out[i] = (__bf16)in[(size_t)r * C + c];
  }
}

// ---- WMMA GEMM with TDM-staged B panels ------------------------------------
// C[M,N] = A[M,K] * Bt[N,K]^T + bias.  Block = 256 thr = 8 waves, 64x128 tile;
// wave = 16x64 (A fragment reused across 4 N-tiles). The 128x32 B panel for
// the whole block is DMA'd into LDS by the TDM (double-buffered): wave0 issues
// the descriptor + s_wait_tensorcnt, barriers publish it to all 8 waves.
// MODE 0: Q (scaled) / 1: K / 2: V^T  -> bf16 head-interleaved
// MODE 3: FFN1 (+bias, exact GELU) -> bf16 ; MODE 4: FFN2 (+bias) -> f32
template <int MODE>
__global__ __launch_bounds__(256) void gemm_wmma(const __bf16* __restrict__ A,
                                                 const __bf16* __restrict__ Bt,
                                                 const float* __restrict__ bias,
                                                 float* __restrict__ outF,
                                                 __bf16* __restrict__ outB,
                                                 int N, int K) {
  __shared__ __attribute__((aligned(16))) __bf16 bpanel[2][128 * 32]; // 16 KB
  const int tid  = threadIdx.x;
  const int lane = tid & 31;
  const int wave = tid >> 5;
  const int nblk = blockIdx.x * 128;
  const int m0   = blockIdx.y * 64 + (wave & 3) * 16;
  const int wn   = wave >> 2;                 // 0..1 -> 64-wide N slice

  v8f acc[4] = {};
  const __bf16* aBase = A + (size_t)m0 * K;
  const __bf16* aLane = aBase + (size_t)(lane & 15) * K + (lane >> 4) * 8;

  if (tid == 0)  // prime the pipeline (TDM ignores EXEC; wave0 issues once)
    tdm_load_panel(Bt + (size_t)nblk * K, &bpanel[0][0], K, N);

  int buf = 0;
  for (int k0 = 0; k0 < K; k0 += 32, buf ^= 1) {
    if (tid == 0) __builtin_amdgcn_s_wait_tensorcnt(0); // panel `buf` landed
    __syncthreads();                                    // publish to all waves
    if (tid == 0 && k0 + 32 < K)                        // overlap next DMA
      tdm_load_panel(Bt + (size_t)nblk * K + (k0 + 32), &bpanel[buf ^ 1][0], K, N);

    __builtin_prefetch(aLane + k0 + 32, 0, 0);          // global_prefetch_b8
    v16bf af = load_frag(aBase + k0, K);
    const __bf16* bb = &bpanel[buf][(size_t)wn * 64 * 32];
#pragma unroll
    for (int t = 0; t < 4; ++t) {
      v16bf bf = load_frag(bb + (size_t)t * 16 * 32, 32);  // ds_load_b128
      acc[t] = wmma_bf16(af, bf, acc[t]);
    }
    __syncthreads();   // all waves done reading `buf` before it is rewritten
  }

  // C/D layout: lane = column n (l&15), VGPR r = row m = 8*(l>>4)+r
  const int half = lane >> 4, n16 = lane & 15;
  const int n0 = nblk + wn * 64;
#pragma unroll
  for (int t = 0; t < 4; ++t) {
#pragma unroll
    for (int r = 0; r < 8; ++r) {
      const int m = m0 + half * 8 + r;
      const int n = n0 + t * 16 + n16;
      float c = acc[t][r] + bias[n];
      if constexpr (MODE == 0) {
        c *= 0.125f;  // DK^-0.5
        int b = m >> 11, s = m & (S - 1), h = n >> 6, d = n & 63;
        outB[((size_t)(b * NH + h) * S + s) * DK + d] = (__bf16)c;
      } else if constexpr (MODE == 1) {
        int b = m >> 11, s = m & (S - 1), h = n >> 6, d = n & 63;
        outB[((size_t)(b * NH + h) * S + s) * DK + d] = (__bf16)c;
      } else if constexpr (MODE == 2) {
        int b = m >> 11, s = m & (S - 1), h = n >> 6, d = n & 63;
        outB[((size_t)(b * NH + h) * DK + d) * S + s] = (__bf16)c;
      } else if constexpr (MODE == 3) {
        float g = 0.5f * c * (1.0f + erff(c * 0.70710678118654752f));
        outB[(size_t)m * N + n] = (__bf16)g;
      } else {
        outF[(size_t)m * N + n] = c;
      }
    }
  }
}

// ---- flash attention: 4 independent 16-query tiles per 128-thread block ----
// Scores: Q[16x64] @ K^T -> two 16x16 tiles per 32-key block (4 WMMAs).
// Online softmax with 16-lane shfl_xor reductions (wave32 halves hold the
// independent row groups m=0..7 / m=8..15, matching the C layout). P is
// staged through a per-wave LDS region to re-layout into an A fragment, then
// ctx[16x64] += P[16x32] @ V[32x64] (4 WMMAs; pre-transposed Vt -> contiguous).
__global__ __launch_bounds__(128) void attn_wmma(const __bf16* __restrict__ Q,
                                                 const __bf16* __restrict__ Kt,
                                                 const __bf16* __restrict__ Vt,
                                                 float* __restrict__ Y) {
  __shared__ __attribute__((aligned(16))) __bf16 lds_p[4][16 * 32];

  const int lane = threadIdx.x & 31;
  const int wave = threadIdx.x >> 5;
  const int tile = blockIdx.x * 4 + wave;     // global q-tile id
  const int qt = tile & (S / 16 - 1);         // S/16 = 128
  const int bh = tile >> 7;
  const int b = bh >> 4, h = bh & (NH - 1);
  const int q0 = qt * 16;
  const int half = lane >> 4, n16 = lane & 15;
  __bf16* pbuf = &lds_p[wave][0];

  const __bf16* qBase = Q + ((size_t)bh * S + q0) * DK;
  const v16bf aq0 = load_frag(qBase, DK);        // k = 0..31
  const v16bf aq1 = load_frag(qBase + 32, DK);   // k = 32..63

  float mrow[8], lrow[8];
#pragma unroll
  for (int r = 0; r < 8; ++r) { mrow[r] = -3.0e38f; lrow[r] = 0.f; }
  v8f acc[4] = {};

  for (int kb = 0; kb < S; kb += 32) {
    const __bf16* kBase = Kt + ((size_t)bh * S + kb) * DK;
    v8f s0 = {}, s1 = {};
    s0 = wmma_bf16(aq0, load_frag(kBase, DK), s0);
    s0 = wmma_bf16(aq1, load_frag(kBase + 32, DK), s0);
    s1 = wmma_bf16(aq0, load_frag(kBase + (size_t)16 * DK, DK), s1);
    s1 = wmma_bf16(aq1, load_frag(kBase + (size_t)16 * DK + 32, DK), s1);

#pragma unroll
    for (int r = 0; r < 8; ++r) {
      float cmax = fmaxf(s0[r], s1[r]);
#pragma unroll
      for (int msk = 1; msk < 16; msk <<= 1)
        cmax = fmaxf(cmax, __shfl_xor(cmax, msk, 32));
      const float mnew = fmaxf(mrow[r], cmax);
      const float corr = __expf(mrow[r] - mnew);
      const float p0 = __expf(s0[r] - mnew);
      const float p1 = __expf(s1[r] - mnew);
      float ps = p0 + p1;
#pragma unroll
      for (int msk = 1; msk < 16; msk <<= 1) ps += __shfl_xor(ps, msk, 32);
      lrow[r] = lrow[r] * corr + ps;
      mrow[r] = mnew;
#pragma unroll
      for (int t = 0; t < 4; ++t) acc[t][r] *= corr;
      const int mr = half * 8 + r;
      pbuf[mr * 32 + n16]      = (__bf16)p0;
      pbuf[mr * 32 + 16 + n16] = (__bf16)p1;
    }
    __syncthreads();  // same trip count for all 4 waves -> legal
    const v16bf pa = load_frag(pbuf, 32);
#pragma unroll
    for (int t = 0; t < 4; ++t) {
      v16bf bv = load_frag(Vt + ((size_t)bh * DK + t * 16) * S + kb, S);
      acc[t] = wmma_bf16(pa, bv, acc[t]);
    }
    __syncthreads();
  }

#pragma unroll
  for (int r = 0; r < 8; ++r) {
    const float inv = 1.0f / lrow[r];
    const int s = q0 + half * 8 + r;
#pragma unroll
    for (int t = 0; t < 4; ++t)
      Y[(size_t)(b * S + s) * H + h * DK + t * 16 + n16] = acc[t][r] * inv;
  }
}

// ---- fused residual add + LayerNorm (f32), optional bf16 mirror ------------
__global__ __launch_bounds__(256) void add_ln(const float* __restrict__ a,
                                              const float* __restrict__ res,
                                              const float* __restrict__ gamma,
                                              const float* __restrict__ beta,
                                              float* __restrict__ outF,
                                              __bf16* __restrict__ outB) {
  const int row = blockIdx.x;
  const int tid = threadIdx.x;
  const float* pa = a + (size_t)row * H;
  const float* pr = res + (size_t)row * H;

  float v[4], s = 0.f, s2 = 0.f;
#pragma unroll
  for (int i = 0; i < 4; ++i) {
    const int c = tid + i * 256;
    v[i] = pa[c] + pr[c];
    s += v[i]; s2 += v[i] * v[i];
  }
#pragma unroll
  for (int msk = 1; msk < 32; msk <<= 1) {
    s  += __shfl_xor(s,  msk, 32);
    s2 += __shfl_xor(s2, msk, 32);
  }
  __shared__ float sh[16];
  const int w = tid >> 5;
  if ((tid & 31) == 0) { sh[w] = s; sh[8 + w] = s2; }
  __syncthreads();
  if (tid == 0) {
    float ts = 0.f, ts2 = 0.f;
    for (int i = 0; i < 8; ++i) { ts += sh[i]; ts2 += sh[8 + i]; }
    sh[0] = ts; sh[8] = ts2;
  }
  __syncthreads();
  const float mean = sh[0] * (1.f / H);
  const float var  = sh[8] * (1.f / H) - mean * mean;
  const float rstd = rsqrtf(var + 1e-5f);
#pragma unroll
  for (int i = 0; i < 4; ++i) {
    const int c = tid + i * 256;
    const float o = (v[i] - mean) * rstd * gamma[c] + beta[c];
    outF[(size_t)row * H + c] = o;
    if (outB) outB[(size_t)row * H + c] = (__bf16)o;
  }
}

// ---------------------------------------------------------------------------
extern "C" void kernel_launch(void* const* d_in, const int* in_sizes, int n_in,
                              void* d_out, int out_size, void* d_ws, size_t ws_size,
                              hipStream_t stream) {
  const float* x   = (const float*)d_in[0];
  const float* wq  = (const float*)d_in[1];
  const float* bq  = (const float*)d_in[2];
  const float* wk  = (const float*)d_in[3];
  const float* bk  = (const float*)d_in[4];
  const float* wv  = (const float*)d_in[5];
  const float* bv  = (const float*)d_in[6];
  const float* w1  = (const float*)d_in[7];
  const float* b1  = (const float*)d_in[8];
  const float* w2  = (const float*)d_in[9];
  const float* b2  = (const float*)d_in[10];
  const float* g1  = (const float*)d_in[11];
  const float* be1 = (const float*)d_in[12];
  const float* g2  = (const float*)d_in[13];
  const float* be2 = (const float*)d_in[14];

  char* ws = (char*)d_ws;
  size_t off = 0;
  auto carve = [&](size_t bytes) -> char* {
    char* p = ws + off;
    off += (bytes + 255) & ~(size_t)255;
    return p;
  };
  __bf16* xb  = (__bf16*)carve((size_t)M * H * 2);
  __bf16* wqT = (__bf16*)carve((size_t)H * H * 2);
  __bf16* wkT = (__bf16*)carve((size_t)H * H * 2);
  __bf16* wvT = (__bf16*)carve((size_t)H * H * 2);
  __bf16* w1T = (__bf16*)carve((size_t)H * FFN * 2);
  __bf16* w2T = (__bf16*)carve((size_t)H * FFN * 2);
  __bf16* qb  = (__bf16*)carve((size_t)M * H * 2);
  __bf16* kb  = (__bf16*)carve((size_t)M * H * 2);
  __bf16* vtb = (__bf16*)carve((size_t)M * H * 2);
  float*  y   = (float*) carve((size_t)M * H * 4);
  float*  y1  = (float*) carve((size_t)M * H * 4);
  __bf16* y1b = (__bf16*)carve((size_t)M * H * 2);
  __bf16* z1  = (__bf16*)carve((size_t)M * FFN * 2);
  float*  z   = (float*) carve((size_t)M * H * 4);
  (void)ws_size; (void)in_sizes; (void)n_in; (void)out_size;

  // 1. cast activations / weights to bf16 (weights transposed to [N,K])
  cast_bf16<<<2048, 256, 0, stream>>>(x, xb, M * H);
  transpose_bf16<<<4096, 256, 0, stream>>>(wq, wqT, H, H);
  transpose_bf16<<<4096, 256, 0, stream>>>(wk, wkT, H, H);
  transpose_bf16<<<4096, 256, 0, stream>>>(wv, wvT, H, H);
  transpose_bf16<<<8192, 256, 0, stream>>>(w1, w1T, H, FFN);
  transpose_bf16<<<8192, 256, 0, stream>>>(w2, w2T, FFN, H);

  // 2. QKV projections (WMMA + TDM staging), head-interleaved bf16 outputs
  gemm_wmma<0><<<dim3(H / 128, M / 64), 256, 0, stream>>>(xb, wqT, bq, nullptr, qb, H, H);
  gemm_wmma<1><<<dim3(H / 128, M / 64), 256, 0, stream>>>(xb, wkT, bk, nullptr, kb, H, H);
  gemm_wmma<2><<<dim3(H / 128, M / 64), 256, 0, stream>>>(xb, wvT, bv, nullptr, vtb, H, H);

  // 3. flash attention (WMMA scores + WMMA PV, online softmax)
  attn_wmma<<<dim3(BB * NH * (S / 16) / 4), dim3(128), 0, stream>>>(qb, kb, vtb, y);

  // 4. post-norm 1: y1 = LN(x + attn)
  add_ln<<<dim3(M), dim3(256), 0, stream>>>(y, x, g1, be1, y1, y1b);

  // 5. FFN: z1 = gelu(y1 @ w1 + b1);  z = z1 @ w2 + b2
  gemm_wmma<3><<<dim3(FFN / 128, M / 64), 256, 0, stream>>>(y1b, w1T, b1, nullptr, z1, FFN, H);
  gemm_wmma<4><<<dim3(H / 128, M / 64), 256, 0, stream>>>(z1, w2T, b2, z, nullptr, H, FFN);

  // 6. post-norm 2 -> output (f32)
  add_ln<<<dim3(M), dim3(256), 0, stream>>>(z, y1, g2, be2, (float*)d_out, nullptr);
}
